// EEG_STFNet_87033217286511
// MI455X (gfx1250) — compile-verified, matrix operand order checked by reading, not run on hardware
//
#include <hip/hip_runtime.h>
#include <hip/hip_bf16.h>

#define BATCH   32
#define NBANDS  9
#define CHANS   22
#define SAMPLES 1125
#define NCLASSES 4
#define UNITS   64
#define FUSED   128

typedef __attribute__((ext_vector_type(16))) _Float16 v16h;
typedef __attribute__((ext_vector_type(8)))  float    v8f;

__device__ __forceinline__ float sigmoidf_(float x) { return 1.f / (1.f + __expf(-x)); }
__device__ __forceinline__ float eluf_(float x)     { return x > 0.f ? x : (__expf(x) - 1.f); }
__device__ __forceinline__ float bnscale_()         { return rsqrtf(1.0f + 1e-5f); }

// ---------------------------------------------------------------------------
// K1: cast fp32 -> f16
// ---------------------------------------------------------------------------
__global__ __launch_bounds__(256) void cast_f16_kernel(const float* __restrict__ in,
                                                       _Float16* __restrict__ out, int n) {
    for (int i = blockIdx.x * 256 + threadIdx.x; i < n; i += gridDim.x * 256)
        out[i] = (_Float16)in[i];
}

// ---------------------------------------------------------------------------
// K2: pack conv1 weights [nb,F1,1,1,KL] -> f16 [nb][16][KL], rows >= F1 zero
// ---------------------------------------------------------------------------
template <int F1, int KL>
__global__ __launch_bounds__(256) void pack_w_kernel(const float* __restrict__ c1w,
                                                     _Float16* __restrict__ wout) {
    int n = NBANDS * 16 * KL;
    for (int i = blockIdx.x * 256 + threadIdx.x; i < n; i += gridDim.x * 256) {
        int band = i / (16 * KL);
        int rem  = i % (16 * KL);
        int m = rem / KL, k = rem % KL;
        float v = (m < F1) ? c1w[(band * F1 + m) * KL + k] : 0.f;
        wout[i] = (_Float16)v;
    }
}

// ---------------------------------------------------------------------------
// K3: fused temporal-conv (WMMA f16) + BN1 + ELU + graph(=mean) + depthwise + BN2
//     one workgroup = one (band, batch, 16-wide time tile); 8 waves
//     output xdw: [NB][B][Cc][T] fp32
// ---------------------------------------------------------------------------
template <int F1, int KL>
__global__ __launch_bounds__(256) void conv_block_kernel(
    const _Float16* __restrict__ x16,  // [B][NB][C][T]
    const _Float16* __restrict__ w16,  // [NB][16][KL]
    const float* __restrict__ c1b, const float* __restrict__ bn1g, const float* __restrict__ bn1b,
    const float* __restrict__ gw,  const float* __restrict__ gb,
    const float* __restrict__ dww, const float* __restrict__ dwb,
    const float* __restrict__ bn2g, const float* __restrict__ bn2b,
    float* __restrict__ xdw) {
    constexpr int Cc  = F1 * 2;
    constexpr int TT  = 16 + KL - 1;
    constexpr int TTP = ((TT + 15) / 16) * 16;
    constexpr int LO  = (KL - 1) / 2;

    __shared__ _Float16 ldsW[16 * KL];
    __shared__ _Float16 ldsX[CHANS][TTP];
    __shared__ float    xt[F1][CHANS][16];
    __shared__ float    sArr[16];

    const int band  = blockIdx.x / BATCH;
    const int b     = blockIdx.x % BATCH;
    const int tbase = blockIdx.y * 16;
    const float rs  = bnscale_();

    // stage weights
    for (int i = threadIdx.x; i < 16 * KL; i += 256)
        ldsW[i] = w16[band * 16 * KL + i];
    // stage x tile with halo + zero pad
    const _Float16* xb = x16 + ((size_t)(b * NBANDS + band) * CHANS) * SAMPLES;
    for (int i = threadIdx.x; i < CHANS * TT; i += 256) {
        int c = i / TT, j = i % TT;
        int t = tbase - LO + j;
        _Float16 v = (_Float16)0.f;
        if (t >= 0 && t < SAMPLES) v = xb[c * SAMPLES + t];
        ldsX[c][j] = v;
    }
    // prefetch next tile (emits global_prefetch_b8)
    {
        int tn = tbase + 16 + threadIdx.x;
        if (tn < SAMPLES) __builtin_prefetch(&xb[tn], 0, 0);
    }
    __syncthreads();

    const int lane = threadIdx.x & 31;
    const int wave = threadIdx.x >> 5;
    const int nIdx = lane & 15;
    const int half = (lane >> 4) & 1;

    for (int c = wave; c < CHANS; c += 8) {
        v8f acc = {0.f, 0.f, 0.f, 0.f, 0.f, 0.f, 0.f, 0.f};
        for (int k0 = 0; k0 < KL; k0 += 32) {
            v16h afrag, bfrag;
#pragma unroll
            for (int e = 0; e < 16; ++e) {
                int r = e >> 1, bit = e & 1;
                int kk = (r < 4) ? (2 * r + bit) : (16 + 2 * (r - 4) + bit);
                kk += half * 8;
                afrag[e] = ldsW[nIdx * KL + k0 + kk];   // A: M=nIdx (row), K striped per ISA
                bfrag[e] = ldsX[c][nIdx + k0 + kk];     // B: N=nIdx (col), sliding window
            }
            acc = __builtin_amdgcn_wmma_f32_16x16x32_f16(
                false, afrag, false, bfrag, (short)0, acc, false, false);
        }
#pragma unroll
        for (int r = 0; r < 8; ++r) {
            int m = r + 8 * half;   // output row M
            if (m < F1) {
                float v = acc[r] + c1b[band * F1 + m];
                v = v * bn1g[band * F1 + m] * rs + bn1b[band * F1 + m];
                xt[m][c][nIdx] = eluf_(v);
            }
        }
    }
    __syncthreads();

    // graph term: AHAT == 1/22 everywhere -> s[t] = mean over (F1, C)
    if (threadIdx.x < 16) {
        float sum = 0.f;
        for (int c = 0; c < CHANS; ++c)
            for (int f = 0; f < F1; ++f) sum += xt[f][c][threadIdx.x];
        sArr[threadIdx.x] = sum / (float)(F1 * CHANS);
    }
    __syncthreads();

    // depthwise over electrodes + BN2
    for (int idx = threadIdx.x; idx < Cc * 16; idx += 256) {
        int ch = idx >> 4, n = idx & 15;
        int f = ch >> 1, dpt = ch & 1;
        float g = sArr[n] * gw[band * F1 + f] + gb[band * F1 + f];
        float a = 0.f;
        for (int c = 0; c < CHANS; ++c)
            a += (xt[f][c][n] + g) * dww[((band * F1 + f) * 2 + dpt) * CHANS + c];
        a += dwb[band * Cc + ch];
        a = a * bn2g[band * Cc + ch] * rs + bn2b[band * Cc + ch];
        int t = tbase + n;
        if (t < SAMPLES)
            xdw[(((size_t)band * BATCH + b) * Cc + ch) * SAMPLES + t] = a;
    }
}

// ---------------------------------------------------------------------------
// K4a: channel attention -> sv, plus channel-mean/max maps c2
// ---------------------------------------------------------------------------
template <int Cc>
__global__ __launch_bounds__(256) void chan_att_kernel(
    const float* __restrict__ xdw,
    const float* __restrict__ ca1w, const float* __restrict__ ca1b,
    const float* __restrict__ ca2w, const float* __restrict__ ca2b,
    float* __restrict__ sv_out, float* __restrict__ c2_out) {
    constexpr int HID = (Cc / 16 > 1) ? (Cc / 16) : 1;
    const int band = blockIdx.x / BATCH;
    const int b    = blockIdx.x % BATCH;
    const float* xp = xdw + ((size_t)(band * BATCH + b) * Cc) * SAMPLES;

    __shared__ float z[Cc];
    __shared__ float sv[Cc];

    if (threadIdx.x < Cc) {
        int ch = threadIdx.x;
        float m = 0.f, mx = -3.4e38f;
        for (int t = 0; t < SAMPLES; ++t) {
            float v = xp[ch * SAMPLES + t];
            m += v; mx = fmaxf(mx, v);
        }
        z[ch] = m / (float)SAMPLES + mx;
    }
    __syncthreads();
    if (threadIdx.x == 0) {
        float h[HID];
        for (int j = 0; j < HID; ++j) {
            float a = ca1b[band * HID + j];
            for (int ch = 0; ch < Cc; ++ch) a += ca1w[(band * HID + j) * Cc + ch] * z[ch];
            h[j] = fmaxf(a, 0.f);
        }
        for (int ch = 0; ch < Cc; ++ch) {
            float a = ca2b[band * Cc + ch];
            for (int j = 0; j < HID; ++j) a += ca2w[(band * Cc + ch) * HID + j] * h[j];
            sv[ch] = sigmoidf_(a);
        }
    }
    __syncthreads();
    if (threadIdx.x < Cc)
        sv_out[(band * BATCH + b) * Cc + threadIdx.x] = sv[threadIdx.x];
    for (int t = threadIdx.x; t < SAMPLES; t += 256) {
        float m = 0.f, mx = -3.4e38f;
        for (int ch = 0; ch < Cc; ++ch) {
            float v = xp[ch * SAMPLES + t] * sv[ch];
            m += v; mx = fmaxf(mx, v);
        }
        c2_out[((size_t)(band * BATCH + b) * 2 + 0) * SAMPLES + t] = m / (float)Cc;
        c2_out[((size_t)(band * BATCH + b) * 2 + 1) * SAMPLES + t] = mx;
    }
}

// ---------------------------------------------------------------------------
// K4b: spatial attention + ELU + pool1 + sep-conv + BN3 + ELU + pool2 -> feats
// ---------------------------------------------------------------------------
template <int Cc, int K4, int FOFF>
__global__ __launch_bounds__(256) void sp_sep_kernel(
    const float* __restrict__ xdw, const float* __restrict__ sv_in,
    const float* __restrict__ c2,
    const float* __restrict__ saw, const float* __restrict__ sabg, const float* __restrict__ sabb,
    const float* __restrict__ sepw, const float* __restrict__ sepb,
    const float* __restrict__ bn3g, const float* __restrict__ bn3b,
    float* __restrict__ feats) {
    constexpr int TP1 = SAMPLES / K4;
    constexpr int T3  = TP1 / K4;
    constexpr int LO2 = (K4 - 1) / 2;
    const int band = blockIdx.x / BATCH;
    const int b    = blockIdx.x % BATCH;
    const float rs = bnscale_();

    __shared__ float asig[SAMPLES];
    __shared__ float xp[Cc][TP1];
    __shared__ float ys[Cc][TP1];
    __shared__ float svl[Cc];

    if (threadIdx.x < Cc)
        svl[threadIdx.x] = sv_in[(band * BATCH + b) * Cc + threadIdx.x];

    const float* c2p = c2 + (size_t)(band * BATCH + b) * 2 * SAMPLES;
    for (int t = threadIdx.x; t < SAMPLES; t += 256) {
        float a = 0.f;
        for (int i = 0; i < 2; ++i)
            for (int d = -1; d <= 1; ++d) {
                int tt = t + d;
                if (tt >= 0 && tt < SAMPLES)
                    a += c2p[i * SAMPLES + tt] * saw[(band * 2 + i) * 9 + 3 + (d + 1)];
            }
        a = a * sabg[band] * rs + sabb[band];
        asig[t] = sigmoidf_(a);
    }
    __syncthreads();

    const float* xq = xdw + (size_t)(band * BATCH + b) * Cc * SAMPLES;
    for (int idx = threadIdx.x; idx < Cc * TP1; idx += 256) {
        int ch = idx / TP1, tp = idx % TP1;
        float s = 0.f;
        for (int i = 0; i < K4; ++i) {
            int t = tp * K4 + i;
            s += eluf_(xq[ch * SAMPLES + t] * svl[ch] * asig[t]);
        }
        xp[ch][tp] = s / (float)K4;
    }
    __syncthreads();

    for (int idx = threadIdx.x; idx < Cc * TP1; idx += 256) {
        int ch = idx / TP1, tp = idx % TP1;
        float a = sepb[band * Cc + ch];
        for (int k = 0; k < K4; ++k) {
            int j = tp - LO2 + k;
            if (j >= 0 && j < TP1) a += xp[ch][j] * sepw[(band * Cc + ch) * K4 + k];
        }
        a = a * bn3g[band * Cc + ch] * rs + bn3b[band * Cc + ch];
        ys[ch][tp] = eluf_(a);
    }
    __syncthreads();

    for (int idx = threadIdx.x; idx < Cc * T3; idx += 256) {
        int ch = idx / T3, t3 = idx % T3;
        float s = 0.f;
        for (int i = 0; i < K4; ++i) s += ys[ch][t3 * K4 + i];
        s /= (float)K4;
        feats[((size_t)b * NBANDS + band) * 96 + FOFF + ch * T3 + t3] = s;
    }
}

// ---------------------------------------------------------------------------
// K5: frequency attention + fusion + BN + ELU + LTC (6 unfolds) + FC
// ---------------------------------------------------------------------------
__global__ __launch_bounds__(256) void head_kernel(
    const float* __restrict__ feats,
    const float* __restrict__ fa1w, const float* __restrict__ fa1b,
    const float* __restrict__ fa2w, const float* __restrict__ fa2b,
    const float* __restrict__ fusw, const float* __restrict__ fusb,
    const float* __restrict__ fbng, const float* __restrict__ fbnb,
    const float* __restrict__ in_w, const float* __restrict__ in_b,
    const float* __restrict__ s_mu, const float* __restrict__ s_sig,
    const float* __restrict__ s_w,  const float* __restrict__ s_erev,
    const float* __restrict__ mu_p, const float* __restrict__ sig_p,
    const float* __restrict__ w_p,  const float* __restrict__ erev_p,
    const float* __restrict__ gleak, const float* __restrict__ vleak,
    const float* __restrict__ cm,
    const float* __restrict__ out_w, const float* __restrict__ out_b,
    const float* __restrict__ fcw,  const float* __restrict__ fcb,
    float* __restrict__ out) {
    __shared__ float att[BATCH][NBANDS];
    __shared__ float xi[BATCH][FUSED];
    __shared__ float numS[BATCH][UNITS], denS[BATCH][UNITS];
    __shared__ float v0[BATCH * UNITS], v1[BATCH * UNITS];
    const float rs = bnscale_();
    const int tid = threadIdx.x;

    if (tid < BATCH) {
        const int b = tid;
        float y[NBANDS];
        for (int nb = 0; nb < NBANDS; ++nb) {
            float s = 0.f;
            for (int f = 0; f < 96; ++f) s += feats[((size_t)b * NBANDS + nb) * 96 + f];
            y[nb] = s / 96.f;
        }
        float h1[NBANDS / 2];
        for (int j = 0; j < NBANDS / 2; ++j) {
            float a = fa1b[j];
            for (int nb = 0; nb < NBANDS; ++nb) a += fa1w[j * NBANDS + nb] * y[nb];
            h1[j] = eluf_(a);
        }
        float l[NBANDS], mx = -3.4e38f;
        for (int nb = 0; nb < NBANDS; ++nb) {
            float a = fa2b[nb];
            for (int j = 0; j < NBANDS / 2; ++j) a += fa2w[nb * (NBANDS / 2) + j] * h1[j];
            l[nb] = a; mx = fmaxf(mx, a);
        }
        float se = 0.f;
        for (int nb = 0; nb < NBANDS; ++nb) { float e = __expf(l[nb] - mx); att[b][nb] = e; se += e; }
        for (int nb = 0; nb < NBANDS; ++nb) att[b][nb] /= se;
    }
    __syncthreads();

    for (int idx = tid; idx < BATCH * FUSED; idx += 256) {
        int b = idx / FUSED, o = idx % FUSED;
        float a = fusb[o];
        for (int k = 0; k < NBANDS * 96; ++k)
            a += feats[(size_t)b * NBANDS * 96 + k] * att[b][k / 96] * fusw[o * NBANDS * 96 + k];
        a = eluf_(a * fbng[o] * rs + fbnb[o]);
        xi[b][o] = a * in_w[o] + in_b[o];
    }
    __syncthreads();

    for (int idx = tid; idx < BATCH * UNITS; idx += 256) {
        int b = idx / UNITS, u = idx % UNITS;
        float ns = 0.f, ds = 0.f;
        for (int i = 0; i < FUSED; ++i) {
            float sg = sigmoidf_((xi[b][i] - s_mu[i * UNITS + u]) * s_sig[i * UNITS + u]);
            float sa = s_w[i * UNITS + u] * sg;
            ns += sa * s_erev[i * UNITS + u];
            ds += sa;
        }
        numS[b][u] = ns; denS[b][u] = ds;
        v0[b * UNITS + u] = 0.f;
    }
    __syncthreads();

    float* va = v0;
    float* vb = v1;
    for (int it = 0; it < 6; ++it) {
        for (int idx = tid; idx < BATCH * UNITS; idx += 256) {
            int b = idx / UNITS, u = idx % UNITS;
            float n = numS[b][u], d = denS[b][u];
            for (int j = 0; j < UNITS; ++j) {
                float sg = sigmoidf_((va[b * UNITS + j] - mu_p[j * UNITS + u]) * sig_p[j * UNITS + u]);
                float ww = w_p[j * UNITS + u] * sg;
                n += ww * erev_p[j * UNITS + u];
                d += ww;
            }
            float cmt = cm[u] * 6.f;
            vb[b * UNITS + u] =
                (cmt * va[b * UNITS + u] + gleak[u] * vleak[u] + n) / (cmt + gleak[u] + d + 1e-8f);
        }
        __syncthreads();
        float* tmp = va; va = vb; vb = tmp;
    }

    for (int idx = tid; idx < BATCH * NCLASSES; idx += 256) {
        int b = idx / NCLASSES, k = idx % NCLASSES;
        float a = fcb[k];
        for (int u = 0; u < UNITS; ++u)
            a += (va[b * UNITS + u] * out_w[u] + out_b[u]) * fcw[k * UNITS + u];
        out[b * NCLASSES + k] = a;
    }
}

// ---------------------------------------------------------------------------
// launch
// ---------------------------------------------------------------------------
extern "C" void kernel_launch(void* const* d_in, const int* in_sizes, int n_in,
                              void* d_out, int out_size, void* d_ws, size_t ws_size,
                              hipStream_t stream) {
    (void)in_sizes; (void)n_in; (void)out_size; (void)ws_size;
    // input indices: 0=x, pa @1..21, pb @22..42, pf @43..67 (dict insertion order)
    const float* x = (const float*)d_in[0];
    const int PA = 1, PB = 22, PF = 43;
    auto F = [&](int i) { return (const float*)d_in[i]; };

    // workspace carve-out
    char* base = (char*)d_ws;
    size_t off = 0;
    auto alloc = [&](size_t bytes) {
        void* p = base + off;
        off = (off + bytes + 255) & ~(size_t)255;
        return p;
    };
    const size_t NX = (size_t)BATCH * NBANDS * CHANS * SAMPLES;
    _Float16* x16  = (_Float16*)alloc(NX * 2);
    _Float16* wA16 = (_Float16*)alloc((size_t)NBANDS * 16 * 64 * 2);
    _Float16* wB16 = (_Float16*)alloc((size_t)NBANDS * 16 * 96 * 2);
    float* xdwA = (float*)alloc((size_t)NBANDS * BATCH * 16 * SAMPLES * 4);
    float* xdwB = (float*)alloc((size_t)NBANDS * BATCH * 32 * SAMPLES * 4);
    float* svA  = (float*)alloc((size_t)NBANDS * BATCH * 16 * 4);
    float* svB  = (float*)alloc((size_t)NBANDS * BATCH * 32 * 4);
    float* c2A  = (float*)alloc((size_t)NBANDS * BATCH * 2 * SAMPLES * 4);
    float* c2B  = (float*)alloc((size_t)NBANDS * BATCH * 2 * SAMPLES * 4);
    float* feats = (float*)alloc((size_t)BATCH * NBANDS * 96 * 4);

    cast_f16_kernel<<<1024, 256, 0, stream>>>(x, x16, (int)NX);
    pack_w_kernel<8, 64><<<36, 256, 0, stream>>>(F(PA + 0), wA16);
    pack_w_kernel<16, 96><<<54, 256, 0, stream>>>(F(PB + 0), wB16);

    dim3 cg(NBANDS * BATCH, (SAMPLES + 15) / 16);
    conv_block_kernel<8, 64><<<cg, 256, 0, stream>>>(
        x16, wA16, F(PA + 1), F(PA + 2), F(PA + 3), F(PA + 4), F(PA + 5),
        F(PA + 6), F(PA + 7), F(PA + 8), F(PA + 9), xdwA);
    conv_block_kernel<16, 96><<<cg, 256, 0, stream>>>(
        x16, wB16, F(PB + 1), F(PB + 2), F(PB + 3), F(PB + 4), F(PB + 5),
        F(PB + 6), F(PB + 7), F(PB + 8), F(PB + 9), xdwB);

    chan_att_kernel<16><<<NBANDS * BATCH, 256, 0, stream>>>(
        xdwA, F(PA + 10), F(PA + 11), F(PA + 12), F(PA + 13), svA, c2A);
    chan_att_kernel<32><<<NBANDS * BATCH, 256, 0, stream>>>(
        xdwB, F(PB + 10), F(PB + 11), F(PB + 12), F(PB + 13), svB, c2B);

    sp_sep_kernel<16, 16, 0><<<NBANDS * BATCH, 256, 0, stream>>>(
        xdwA, svA, c2A, F(PA + 14), F(PA + 15), F(PA + 16),
        F(PA + 17), F(PA + 18), F(PA + 19), F(PA + 20), feats);
    sp_sep_kernel<32, 24, 64><<<NBANDS * BATCH, 256, 0, stream>>>(
        xdwB, svB, c2B, F(PB + 14), F(PB + 15), F(PB + 16),
        F(PB + 17), F(PB + 18), F(PB + 19), F(PB + 20), feats);

    head_kernel<<<1, 256, 0, stream>>>(
        feats,
        F(PF + 0), F(PF + 1), F(PF + 2), F(PF + 3), F(PF + 4), F(PF + 5),
        F(PF + 6), F(PF + 7), F(PF + 8), F(PF + 9), F(PF + 10), F(PF + 11),
        F(PF + 12), F(PF + 13), F(PF + 14), F(PF + 15), F(PF + 16), F(PF + 17),
        F(PF + 18), F(PF + 19), F(PF + 20), F(PF + 21), F(PF + 22), F(PF + 23),
        F(PF + 24), (float*)d_out);
}
